// CrossAttention3D_70927089926204
// MI455X (gfx1250) — matrix operand order repaired and checked by reading, not compiled, not run on hardware
//
#include <hip/hip_runtime.h>

typedef __attribute__((ext_vector_type(16))) _Float16 v16h;
typedef __attribute__((ext_vector_type(8)))  _Float16 v8h;
typedef __attribute__((ext_vector_type(8)))  float    v8f;

#define NH   12
#define HD   64
#define DIM  768
#define SEQ  1568
#define BATCH 2
#define NCH  24          // 768/32 K-chunks for projections
#define KTILES 49        // 1568/32
#define OT   48          // 768/16 output col tiles
#define MT   196         // 3136/16 row tiles

__device__ __forceinline__ v8f wmma_f16(v16h a, v16h b, v8f c) {
  return __builtin_amdgcn_wmma_f32_16x16x32_f16(false, a, false, b, (short)0, c,
                                                false, false);
}

// ---------------- projection: fp32 X @ W^T + b -> f16, head layouts ---------
__global__ __launch_bounds__(128)
void proj_f32_to_f16(const float* __restrict__ X, const float* __restrict__ W,
                     const float* __restrict__ bias, _Float16* __restrict__ out,
                     int transpose_out) {
  int wave = blockIdx.x * 4 + (threadIdx.x >> 5);
  int lane = threadIdx.x & 31;
  int mt = wave / OT;
  int ot = wave % OT;
  int col = lane & 15;
  int hi  = lane >> 4;
  int row = mt * 16 + col;        // A-matrix row this lane supplies
  int o   = ot * 16 + col;        // output column (C/D layout N = lane&15)
  const float* xr = X + (size_t)row * DIM;
  const float* wr = W + (size_t)o * DIM;
  v8f acc = {};
  for (int ch = 0; ch < NCH; ++ch) {
    int kb = ch * 32;
    v16h a, b;
#pragma unroll
    for (int i = 0; i < 8; ++i) {
      a[i]     = (_Float16)xr[kb + hi * 16 + i];
      a[i + 8] = (_Float16)xr[kb + hi * 16 + 8 + i];
    }
#pragma unroll
    for (int i = 0; i < 16; ++i)
      b[i] = (_Float16)wr[kb + hi * 16 + i];
    acc = wmma_f16(a, b, acc);
  }
  float bv = bias[o];
  int n = o >> 6;          // head
  int c = o & 63;          // channel within head
  // a 16-row tile never straddles the batch boundary (3136 = 2*98*16)
  int bb    = (mt * 16) >= SEQ ? 1 : 0;
  int rbase = mt * 16 - bb * SEQ + 8 * hi;
  size_t obase = transpose_out
      ? ((((size_t)bb * NH + n) * HD + c) * SEQ + rbase)   // vt: [b,n,c,K]
      : ((((size_t)bb * NH + n) * SEQ + rbase) * HD + c);  // qh/kh: [b,n,K,c]
  size_t ostride = transpose_out ? 1 : HD;
#pragma unroll
  for (int j = 0; j < 8; ++j)
    out[obase + (size_t)j * ostride] = (_Float16)(acc[j] + bv);
}

// ---------------- output projection: f16 X @ Wo^T + bo -> fp32 --------------
__global__ __launch_bounds__(128)
void proj_f16_to_f32(const _Float16* __restrict__ X, const float* __restrict__ W,
                     const float* __restrict__ bias, float* __restrict__ out) {
  int wave = blockIdx.x * 4 + (threadIdx.x >> 5);
  int lane = threadIdx.x & 31;
  int mt = wave / OT;
  int ot = wave % OT;
  int col = lane & 15;
  int hi  = lane >> 4;
  int row = mt * 16 + col;
  int o   = ot * 16 + col;
  const _Float16* xr = X + (size_t)row * DIM;
  const float*    wr = W + (size_t)o * DIM;
  v8f acc = {};
  for (int ch = 0; ch < NCH; ++ch) {
    int kb = ch * 32;
    v8h lo = *(const v8h*)(xr + kb + hi * 16);
    v8h hh = *(const v8h*)(xr + kb + hi * 16 + 8);
    v16h a, b;
#pragma unroll
    for (int i = 0; i < 8; ++i) { a[i] = lo[i]; a[i + 8] = hh[i]; }
#pragma unroll
    for (int i = 0; i < 16; ++i)
      b[i] = (_Float16)wr[kb + hi * 16 + i];
    acc = wmma_f16(a, b, acc);
  }
  float bv = bias[o];
  float* orow = out + (size_t)(mt * 16 + 8 * hi) * DIM + o;
#pragma unroll
  for (int j = 0; j < 8; ++j)
    orow[(size_t)j * DIM] = acc[j] + bv;
}

// ---------------- fused flash attention with decomposed 3D rel-pos ----------
__global__ __launch_bounds__(128)
void attn3d(const _Float16* __restrict__ qh, const _Float16* __restrict__ kh,
            const _Float16* __restrict__ vt,
            const float* __restrict__ rpt, const float* __restrict__ rph,
            const float* __restrict__ rpw, _Float16* __restrict__ ao) {
  __shared__ float smem[4 * 3712];          // 59392 B, 4 waves/block
  int wv   = threadIdx.x >> 5;
  int lane = threadIdx.x & 31;
  int tile = blockIdx.x * 4 + wv;
  int qt = tile % 98;
  int n  = (tile / 98) % NH;
  int b  = tile / (98 * NH);

  float*    base = smem + wv * 3712;
  float*    rt   = base;                 // [16][8]   rel_t per (qrow, kt)
  float*    rhw  = base + 128;           // [16][196] rel_h+rel_w per (qrow, khw)
  float*    scr  = base + 128 + 3136;    // 448 floats: rh[16][14], rw[16][14]
  _Float16* pst  = (_Float16*)scr;       // later reused: P stage [16][32] f16

  size_t bn = (size_t)b * NH + n;
  const _Float16* qp = qh + (bn * SEQ + (size_t)qt * 16) * HD;
  const _Float16* kp = kh + bn * SEQ * HD;
  const _Float16* vp = vt + bn * HD * SEQ;

  // --- phase 1: rel-pos dot products (16 rows x (8+14+14) entries) ---
  for (int idx = lane; idx < 576; idx += 32) {
    int qr   = idx / 36;
    int slot = idx % 36;
    int qg   = qt * 16 + qr;
    const _Float16* qrow = qp + (size_t)qr * HD;
    const float* rp; int ridx; float* dst;
    if (slot < 8) {                       // rel_t: Rt[tq][kt] = rpt[tq-kt+7]
      ridx = qg / 196 - slot + 7;
      rp = rpt; dst = &rt[qr * 8 + slot];
    } else if (slot < 22) {               // rel_h: rph[hq-kh+13]
      int kk = slot - 8;
      ridx = (qg / 14) % 14 - kk + 13;
      rp = rph; dst = &scr[qr * 14 + kk];
    } else {                              // rel_w: rpw[wq-kw+13]
      int kk = slot - 22;
      ridx = qg % 14 - kk + 13;
      rp = rpw; dst = &scr[224 + qr * 14 + kk];
    }
    const float* rrow = rp + (size_t)ridx * HD;
    float s = 0.f;
    for (int c = 0; c < HD; ++c) s += (float)qrow[c] * rrow[c];
    *dst = s;
  }
  asm volatile("s_wait_dscnt 0" ::: "memory");
  for (int e = lane; e < 3136; e += 32) { // rhw[qr][khw] = rh[kh] + rw[kw]
    int qr = e / 196, khw = e % 196;
    rhw[e] = scr[qr * 14 + khw / 14] + scr[224 + qr * 14 + khw % 14];
  }
  asm volatile("s_wait_dscnt 0" ::: "memory");

  int col = lane & 15;
  int hi  = lane >> 4;

  // Q tile A-fragments, two K-chunks over head_dim 64
  v16h aq0, aq1;
  {
    const _Float16* qrow = qp + (size_t)col * HD;
    v8h l0 = *(const v8h*)(qrow + hi * 16);
    v8h h0 = *(const v8h*)(qrow + hi * 16 + 8);
    v8h l1 = *(const v8h*)(qrow + 32 + hi * 16);
    v8h h1 = *(const v8h*)(qrow + 40 + hi * 16);
#pragma unroll
    for (int i = 0; i < 8; ++i) {
      aq0[i] = l0[i]; aq0[i + 8] = h0[i];
      aq1[i] = l1[i]; aq1[i + 8] = h1[i];
    }
  }

  const float scale = 0.125f;                 // 1/sqrt(64)
  const float L2E   = 1.4426950408889634f;
  float m0[8], lsum[8];
  v8f o0 = {}, o1 = {}, o2 = {}, o3 = {};
#pragma unroll
  for (int j = 0; j < 8; ++j) { m0[j] = -1e30f; lsum[j] = 0.f; }

  for (int kt32 = 0; kt32 < KTILES; ++kt32) {
    int kb = kt32 * 32;
    if (kt32 + 1 < KTILES) {                 // prefetch next tile (L2-resident)
      __builtin_prefetch(kp + (size_t)(kb + 32 + col) * HD, 0, 3);
      __builtin_prefetch(vp + (size_t)col * SEQ + kb + 32, 0, 3);
    }
    // S = Q * K^T : two 16-col subtiles, chained over c (2 chunks of 32)
    v8f s0 = {}, s1 = {};
    {
      const _Float16* kr0 = kp + (size_t)(kb + col) * HD;
      v16h b00 = *(const v16h*)(kr0 + hi * 16);
      v16h b01 = *(const v16h*)(kr0 + 32 + hi * 16);
      s0 = wmma_f16(aq0, b00, s0);
      s0 = wmma_f16(aq1, b01, s0);
      const _Float16* kr1 = kr0 + (size_t)16 * HD;
      v16h b10 = *(const v16h*)(kr1 + hi * 16);
      v16h b11 = *(const v16h*)(kr1 + 32 + hi * 16);
      s1 = wmma_f16(aq0, b10, s1);
      s1 = wmma_f16(aq1, b11, s1);
    }
    int k0 = kb + col, k1 = k0 + 16;
    int t0 = k0 / 196, hw0 = k0 % 196;
    int t1 = k1 / 196, hw1 = k1 % 196;
#pragma unroll
    for (int j = 0; j < 8; ++j) {
      int qr = j + 8 * hi;
      float f0 = s0[j] * scale + rt[qr * 8 + t0] + rhw[qr * 196 + hw0];
      float f1 = s1[j] * scale + rt[qr * 8 + t1] + rhw[qr * 196 + hw1];
      float mx = fmaxf(f0, f1);
#pragma unroll
      for (int d = 1; d < 16; d <<= 1)        // per-row max across 16 lanes
        mx = fmaxf(mx, __shfl_xor(mx, d, 32));
      float mn   = fmaxf(m0[j], mx);
      float corr = exp2f((m0[j] - mn) * L2E);
      m0[j] = mn;
      float p0 = exp2f((f0 - mn) * L2E);
      float p1 = exp2f((f1 - mn) * L2E);
      lsum[j] = lsum[j] * corr + p0 + p1;
      o0[j] *= corr; o1[j] *= corr; o2[j] *= corr; o3[j] *= corr;
      pst[qr * 32 + col]      = (_Float16)p0;  // stage P (D-layout -> LDS)
      pst[qr * 32 + 16 + col] = (_Float16)p1;
    }
    asm volatile("s_wait_dscnt 0" ::: "memory");
    v16h pa;                                   // reload P in A-layout
    {
      v8h plo = *(const v8h*)(pst + col * 32 + hi * 16);
      v8h phi = *(const v8h*)(pst + col * 32 + hi * 16 + 8);
#pragma unroll
      for (int i = 0; i < 8; ++i) { pa[i] = plo[i]; pa[i + 8] = phi[i]; }
    }
    const _Float16* vb = vp + (size_t)col * SEQ + kb + hi * 16;
    v16h bv0 = *(const v16h*)(vb);
    v16h bv1 = *(const v16h*)(vb + (size_t)16 * SEQ);
    v16h bv2 = *(const v16h*)(vb + (size_t)32 * SEQ);
    v16h bv3 = *(const v16h*)(vb + (size_t)48 * SEQ);
    o0 = wmma_f16(pa, bv0, o0);
    o1 = wmma_f16(pa, bv1, o1);
    o2 = wmma_f16(pa, bv2, o2);
    o3 = wmma_f16(pa, bv3, o3);
  }

  // finalize: reduce row sums, normalize, store f16 [b, q, n*64+c]
#pragma unroll
  for (int j = 0; j < 8; ++j) {
    float ls = lsum[j];
#pragma unroll
    for (int d = 1; d < 16; d <<= 1) ls += __shfl_xor(ls, d, 32);
    float inv = 1.0f / ls;
    int qg = qt * 16 + j + 8 * hi;
    _Float16* orow = ao + ((size_t)b * SEQ + qg) * DIM + n * HD;
    orow[col]      = (_Float16)(o0[j] * inv);
    orow[16 + col] = (_Float16)(o1[j] * inv);
    orow[32 + col] = (_Float16)(o2[j] * inv);
    orow[48 + col] = (_Float16)(o3[j] * inv);
  }
}

extern "C" void kernel_launch(void* const* d_in, const int* in_sizes, int n_in,
                              void* d_out, int out_size, void* d_ws, size_t ws_size,
                              hipStream_t stream) {
  (void)in_sizes; (void)n_in; (void)out_size; (void)ws_size;
  const float* q   = (const float*)d_in[0];
  const float* k   = (const float*)d_in[1];
  const float* Wq  = (const float*)d_in[2];
  const float* bq  = (const float*)d_in[3];
  const float* Wk  = (const float*)d_in[4];
  const float* bk  = (const float*)d_in[5];
  const float* Wv  = (const float*)d_in[6];
  const float* bv  = (const float*)d_in[7];
  const float* Wo  = (const float*)d_in[8];
  const float* bo  = (const float*)d_in[9];
  const float* rpt = (const float*)d_in[10];
  const float* rph = (const float*)d_in[11];
  const float* rpw = (const float*)d_in[12];

  char* ws = (char*)d_ws;
  size_t headbytes = (size_t)BATCH * NH * SEQ * HD * sizeof(_Float16); // 4.82 MB
  _Float16* qh = (_Float16*)(ws);
  _Float16* kh = (_Float16*)(ws + headbytes);
  _Float16* vt = (_Float16*)(ws + 2 * headbytes);
  _Float16* ao = (_Float16*)(ws + 3 * headbytes);

  dim3 blk(128);
  int projBlocks = (MT * OT) / 4;                 // 2352
  proj_f32_to_f16<<<projBlocks, blk, 0, stream>>>(q, Wq, bq, qh, 0);
  proj_f32_to_f16<<<projBlocks, blk, 0, stream>>>(k, Wk, bk, kh, 0);
  proj_f32_to_f16<<<projBlocks, blk, 0, stream>>>(k, Wv, bv, vt, 1);
  int attnBlocks = (BATCH * NH * 98) / 4;         // 588
  attn3d<<<attnBlocks, blk, 0, stream>>>(qh, kh, vt, rpt, rph, rpw, ao);
  proj_f16_to_f32<<<projBlocks, blk, 0, stream>>>(ao, Wo, bo, (float*)d_out);
}
